// RNN_15367392985564
// MI455X (gfx1250) — matrix operand order repaired and checked by reading, not compiled, not run on hardware
//
#include <hip/hip_runtime.h>
#include <hip/hip_bf16.h>

// ---------------------------------------------------------------------------
// Types for WMMA (CDNA5 gfx1250, wave32)
// ---------------------------------------------------------------------------
typedef __attribute__((ext_vector_type(16))) _Float16 v16h;
typedef __attribute__((ext_vector_type(8)))  _Float16 v8h;
typedef __attribute__((ext_vector_type(8)))  float    v8f;

union V16U { v16h v; v8h h[2]; };

// A-matrix fragment (16x32, f16) from row-major storage.
// ISA layout: lane L (m=L&15, hf=L>>4) holds K = hf*8 + [0..7] and 16+hf*8 + [0..7]
// -> two contiguous 16B loads per lane.
__device__ __forceinline__ v16h fragA(const _Float16* base, int pitch, int lane) {
  int m = lane & 15, hf = lane >> 4;
  const _Float16* p = base + (size_t)m * pitch + hf * 8;
  V16U u;
  u.h[0] = *(const v8h*)(p);
  u.h[1] = *(const v8h*)(p + 16);
  return u.v;
}

// B-matrix fragment (32x16, f16) from Bt[n][k] storage (== row-major [out][in] weights).
// ISA layout: lane L (n=L&15, hf=L>>4) holds K = hf*8 + [0..7] and 16+hf*8 + [0..7].
__device__ __forceinline__ v16h fragB(const _Float16* base, int pitch, int lane) {
  int n = lane & 15, hf = lane >> 4;
  const _Float16* p = base + (size_t)n * pitch + hf * 8;
  V16U u;
  u.h[0] = *(const v8h*)(p);
  u.h[1] = *(const v8h*)(p + 16);
  return u.v;
}

__device__ __forceinline__ v8f wmma_f16(v16h a, v16h b, v8f c) {
  return __builtin_amdgcn_wmma_f32_16x16x32_f16(false, a, false, b, (short)0, c,
                                                false, false);
}

__device__ __forceinline__ float fsig(float x)  { return 1.0f / (1.0f + __expf(-x)); }
__device__ __forceinline__ float ftanh(float x) { return 2.0f / (1.0f + __expf(-2.0f * x)) - 1.0f; }

#define NB   64
#define TT   1024
#define INW  64
#define H1   512
#define H2W  256
#define H4W  128

// ---------------------------------------------------------------------------
// Utility kernels
// ---------------------------------------------------------------------------
__global__ void k_f32_to_f16(_Float16* dst, const float* src, int n) {
  int i = blockIdx.x * 256 + threadIdx.x;
  if (i < n) dst[i] = (_Float16)src[i];
}
__global__ void k_zero_f16(_Float16* dst, int n) {
  int i = blockIdx.x * 256 + threadIdx.x;
  if (i < n) dst[i] = (_Float16)0.0f;
}
__global__ void k_zero_f32(float* dst, int n) {
  int i = blockIdx.x * 256 + threadIdx.x;
  if (i < n) dst[i] = 0.0f;
}
__global__ void k_vecadd(float* dst, const float* a, const float* b, int n) {
  int i = blockIdx.x * 256 + threadIdx.x;
  if (i < n) dst[i] = a[i] + b[i];
}
// fc2_w [51][128] -> padded [64][128] f16 (rows 51..63 zero)
__global__ void k_pad_fc2(_Float16* dst, const float* src) {
  int i = blockIdx.x * 256 + threadIdx.x;
  if (i < 64 * 128) dst[i] = (i < 51 * 128) ? (_Float16)src[i] : (_Float16)0.0f;
}

// ---------------------------------------------------------------------------
// Fused LSTM step: gates = [x_t | h_{t-1}] @ [Wx | Wh]^T + b, then cell update.
// One wave owns a (16-batch x 16-hidden) block and all 4 gates for it, so the
// sigmoid/tanh/cell update fuses into the same wave (1 launch per step).
// ---------------------------------------------------------------------------
struct LstmArgs {
  const _Float16* xA;   // A source, x part: [64 (pitch xPitch)][KX] at step t
  const _Float16* hA;   // A source, h part: h_{t-1}, [64 (pitch hPitch)][H]
  const _Float16* Wx;   // [4H][KX] f16 row-major
  const _Float16* Wh;   // [4H][H]  f16 row-major
  const float*    bsum; // [4H] = b_ih + b_hh
  float*          cbuf;    // [64][H] f32 cell state (lives in d_out region)
  float*          hbuf32;  // [64][H] f32 hidden     (lives in d_out region)
  _Float16*       hOut;    // f16 hidden at step t (row pitch hOutPitch)
  int xPitch, KX, hPitch, H, hOutPitch;
};

__device__ __forceinline__ void lstm_step_body(const LstmArgs& A, int blockBase) {
  int lane = threadIdx.x & 31;
  int wid  = (blockIdx.x - blockBase) * 8 + (threadIdx.x >> 5);
  int mt = wid & 3;        // batch tile (4 x 16 = 64)
  int jt = wid >> 2;       // hidden tile
  int j0 = jt * 16;
  int H = A.H, KX = A.KX;

  v8f acc[4] = {};

  const _Float16* xbase = A.xA + (size_t)(mt * 16) * A.xPitch;
  for (int kw = 0; kw < KX / 32; ++kw) {
    v16h a = fragA(xbase + kw * 32, A.xPitch, lane);
#pragma unroll
    for (int g = 0; g < 4; ++g) {
      v16h b = fragB(A.Wx + (size_t)(g * H + j0) * KX + kw * 32, KX, lane);
      acc[g] = wmma_f16(a, b, acc[g]);
    }
  }
  const _Float16* hbase = A.hA + (size_t)(mt * 16) * A.hPitch;
  for (int kw = 0; kw < H / 32; ++kw) {
    v16h a = fragA(hbase + kw * 32, A.hPitch, lane);
#pragma unroll
    for (int g = 0; g < 4; ++g) {
      v16h b = fragB(A.Wh + (size_t)(g * H + j0) * H + kw * 32, H, lane);
      acc[g] = wmma_f16(a, b, acc[g]);
    }
  }

  int hf = lane >> 4;
  int j  = j0 + (lane & 15);
  float b_i = A.bsum[0 * H + j], b_f = A.bsum[1 * H + j];
  float b_g = A.bsum[2 * H + j], b_o = A.bsum[3 * H + j];
#pragma unroll
  for (int r = 0; r < 8; ++r) {
    int m = mt * 16 + r + 8 * hf;           // batch row (C-layout: M = r + 8*half)
    float gi = fsig (acc[0][r] + b_i);
    float gf = fsig (acc[1][r] + b_f);
    float gg = ftanh(acc[2][r] + b_g);
    float go = fsig (acc[3][r] + b_o);
    size_t idx = (size_t)m * H + j;
    float c = gf * A.cbuf[idx] + gi * gg;
    A.cbuf[idx] = c;
    float h = go * ftanh(c);
    A.hbuf32[idx] = h;
    A.hOut[(size_t)m * A.hOutPitch + j] = (_Float16)h;
  }
}

// Fused launch: blocks [0, nb1) run args1 (layer-1 step t+1),
// blocks [nb1, grid) run args2 (layer-2 step t). The two are independent:
// both consume hs1[:,t,:] written by the PREVIOUS launch. This halves the
// sequential launch count of the latency-bound recurrence (2048 -> 1025).
__global__ void __launch_bounds__(256)
k_lstm_pair(LstmArgs a1, int nb1, LstmArgs a2) {
  if ((int)blockIdx.x < nb1) lstm_step_body(a1, 0);
  else                       lstm_step_body(a2, nb1);
}

// ---------------------------------------------------------------------------
// Causal unscaled attention, flash-style online softmax.
// Block: 8 waves; wave w handles query tile qt = qg*8+w (16 query rows).
// kv blocks of 32 keys; V^T staged in LDS (padded pitch), P staged in LDS.
// ---------------------------------------------------------------------------
#define VTP 40   // LDS pitch in halves (mult of 8 -> 16B-aligned b128 frags)

__global__ void __launch_bounds__(256)
k_attn(const _Float16* __restrict__ hs2, _Float16* __restrict__ ctx) {
  __shared__ _Float16 Vt[256 * VTP];       // V^T: Vt[h][s], 32 keys
  __shared__ _Float16 Pb[8][16 * VTP];     // per-wave P tile 16x32

  int lane = threadIdx.x & 31, w = threadIdx.x >> 5;
  int n  = blockIdx.x >> 3;
  int qg = blockIdx.x & 7;
  int qt = qg * 8 + w;
  int q0 = qt * 16;
  int hf = lane >> 4, ln = lane & 15;

  const _Float16* base = hs2 + (size_t)n * TT * H2W;

  v8f O[16];
#pragma unroll
  for (int i = 0; i < 16; ++i) O[i] = {};
  float mrow[8], lrow[8];
#pragma unroll
  for (int r = 0; r < 8; ++r) { mrow[r] = -1e30f; lrow[r] = 0.0f; }

  int kvmax = qg * 4 + 4;   // 32-wide kv windows covering rows up to qg*128+127
  for (int kv = 0; kv < kvmax; ++kv) {
    int s0 = kv * 32;
    { // cooperative stage of V^T (coalesced global b128 reads, strided LDS writes)
      int s  = threadIdx.x >> 3;            // 0..31
      int h0 = (threadIdx.x & 7) * 32;      // 0..224
      const _Float16* src = base + (size_t)(s0 + s) * H2W + h0;
#pragma unroll
      for (int c4 = 0; c4 < 4; ++c4) {
        v8h vv = *(const v8h*)(src + c4 * 8);
#pragma unroll
        for (int e = 0; e < 8; ++e) Vt[(h0 + c4 * 8 + e) * VTP + s] = vv[e];
      }
    }
    __syncthreads();

    if (s0 <= q0 + 15) {                    // this wave still has causal work
      v8f S0 = {}, S1 = {};
      const _Float16* qb = base + (size_t)q0 * H2W;
      const _Float16* kb = base + (size_t)s0 * H2W;
#pragma unroll
      for (int kw = 0; kw < 8; ++kw) {      // K = 256 = 8 * 32
        v16h a  = fragA(qb + kw * 32, H2W, lane);
        v16h b0 = fragB(kb + kw * 32, H2W, lane);
        v16h b1 = fragB(kb + (size_t)16 * H2W + kw * 32, H2W, lane);
        S0 = wmma_f16(a, b0, S0);
        S1 = wmma_f16(a, b1, S1);
      }
      float scl[8];
#pragma unroll
      for (int r = 0; r < 8; ++r) {
        int ig = q0 + r + 8 * hf;
        float s0v = (s0 + ln      <= ig) ? S0[r] : -1e30f;
        float s1v = (s0 + 16 + ln <= ig) ? S1[r] : -1e30f;
        float mx = fmaxf(s0v, s1v);
        mx = fmaxf(mx, __shfl_xor(mx, 1, 32));
        mx = fmaxf(mx, __shfl_xor(mx, 2, 32));
        mx = fmaxf(mx, __shfl_xor(mx, 4, 32));
        mx = fmaxf(mx, __shfl_xor(mx, 8, 32));
        float mnew = fmaxf(mrow[r], mx);
        float sc   = __expf(mrow[r] - mnew);
        float p0   = __expf(s0v - mnew);
        float p1   = __expf(s1v - mnew);
        float rs = p0 + p1;
        rs += __shfl_xor(rs, 1, 32);
        rs += __shfl_xor(rs, 2, 32);
        rs += __shfl_xor(rs, 4, 32);
        rs += __shfl_xor(rs, 8, 32);
        lrow[r] = lrow[r] * sc + rs;
        mrow[r] = mnew;
        scl[r]  = sc;
        Pb[w][(r + 8 * hf) * VTP + ln]      = (_Float16)p0;
        Pb[w][(r + 8 * hf) * VTP + 16 + ln] = (_Float16)p1;
      }
#pragma unroll
      for (int nt = 0; nt < 16; ++nt)
#pragma unroll
        for (int r = 0; r < 8; ++r) O[nt][r] *= scl[r];

      v16h pa = fragA(&Pb[w][0], VTP, lane);
#pragma unroll
      for (int nt = 0; nt < 16; ++nt) {
        v16h b = fragB(&Vt[nt * 16 * VTP], VTP, lane);
        O[nt] = wmma_f16(pa, b, O[nt]);
      }
    }
    __syncthreads();
  }

#pragma unroll
  for (int nt = 0; nt < 16; ++nt)
#pragma unroll
    for (int r = 0; r < 8; ++r) {
      float v = O[nt][r] / lrow[r];
      ctx[((size_t)n * TT + q0 + r + 8 * hf) * H2W + nt * 16 + ln] = (_Float16)v;
    }
}

// ---------------------------------------------------------------------------
// fc1: out1 = ReLU([ctx | hs2] @ fc1_w^T + b).  M=65536, N=128, K=512.
// Wave owns 16x64 strip (4 accumulators).
// ---------------------------------------------------------------------------
__global__ void __launch_bounds__(256)
k_fc1(const _Float16* __restrict__ ctx, const _Float16* __restrict__ hs2,
      const _Float16* __restrict__ W, const float* __restrict__ bias,
      _Float16* __restrict__ out) {
  int lane = threadIdx.x & 31;
  int wid  = blockIdx.x * 8 + (threadIdx.x >> 5);
  int mt = wid >> 1, ns = wid & 1, n0 = ns * 64;
  v8f acc[4] = {};
  const _Float16* ac = ctx + (size_t)mt * 16 * H2W;
  const _Float16* ah = hs2 + (size_t)mt * 16 * H2W;
  for (int kw = 0; kw < 16; ++kw) {
    const _Float16* asrc = (kw < 8) ? (ac + kw * 32) : (ah + (kw - 8) * 32);
    v16h a = fragA(asrc, H2W, lane);
#pragma unroll
    for (int s = 0; s < 4; ++s) {
      v16h b = fragB(W + (size_t)(n0 + s * 16) * 512 + kw * 32, 512, lane);
      acc[s] = wmma_f16(a, b, acc[s]);
    }
  }
  int hf = lane >> 4, ln = lane & 15;
#pragma unroll
  for (int s = 0; s < 4; ++s) {
    int col = n0 + s * 16 + ln;
    float bv = bias[col];
#pragma unroll
    for (int r = 0; r < 8; ++r) {
      int m = mt * 16 + r + 8 * hf;
      float v = fmaxf(acc[s][r] + bv, 0.0f);
      out[(size_t)m * H4W + col] = (_Float16)v;
    }
  }
}

// ---------------------------------------------------------------------------
// fc2: out2 = ReLU(out1 @ fc2_w^T + b).  M=65536, N=64 (51 valid), K=128.
// ---------------------------------------------------------------------------
__global__ void __launch_bounds__(256)
k_fc2(const _Float16* __restrict__ in, const _Float16* __restrict__ W,
      const float* __restrict__ bias, _Float16* __restrict__ out) {
  int lane = threadIdx.x & 31;
  int mt = blockIdx.x * 8 + (threadIdx.x >> 5);
  v8f acc[4] = {};
  const _Float16* ab = in + (size_t)mt * 16 * H4W;
  for (int kw = 0; kw < 4; ++kw) {
    v16h a = fragA(ab + kw * 32, H4W, lane);
#pragma unroll
    for (int s = 0; s < 4; ++s) {
      v16h b = fragB(W + (size_t)(s * 16) * H4W + kw * 32, H4W, lane);
      acc[s] = wmma_f16(a, b, acc[s]);
    }
  }
  int hf = lane >> 4, ln = lane & 15;
#pragma unroll
  for (int s = 0; s < 4; ++s) {
    int col = s * 16 + ln;
    float bv = (col < 51) ? bias[col] : 0.0f;
#pragma unroll
    for (int r = 0; r < 8; ++r) {
      int m = mt * 16 + r + 8 * hf;
      float v = (col < 51) ? fmaxf(acc[s][r] + bv, 0.0f) : 0.0f;
      out[(size_t)m * 64 + col] = (_Float16)v;
    }
  }
}

// fc3: out[m] = sum_{k<51} out2[m][k]*w[k] + b  (tiny, scalar)
__global__ void k_fc3(const _Float16* __restrict__ in, const float* __restrict__ w,
                      const float* __restrict__ b, float* __restrict__ out) {
  int m = blockIdx.x * 256 + threadIdx.x;
  if (m >= NB * TT) return;
  float acc = b[0];
  const _Float16* row = in + (size_t)m * 64;
#pragma unroll 17
  for (int k = 0; k < 51; ++k) acc += (float)row[k] * w[k];
  out[m] = acc;
}

// ---------------------------------------------------------------------------
// Host launcher
// ---------------------------------------------------------------------------
extern "C" void kernel_launch(void* const* d_in, const int* in_sizes, int n_in,
                              void* d_out, int out_size, void* d_ws, size_t ws_size,
                              hipStream_t stream) {
  (void)in_sizes; (void)n_in; (void)out_size; (void)ws_size;
  const float* x      = (const float*)d_in[0];
  const float* W_ih1  = (const float*)d_in[1];
  const float* W_hh1  = (const float*)d_in[2];
  const float* b_ih1  = (const float*)d_in[3];
  const float* b_hh1  = (const float*)d_in[4];
  const float* W_ih2  = (const float*)d_in[5];
  const float* W_hh2  = (const float*)d_in[6];
  const float* b_ih2  = (const float*)d_in[7];
  const float* b_hh2  = (const float*)d_in[8];
  const float* fc1_w  = (const float*)d_in[9];
  const float* fc1_b  = (const float*)d_in[10];
  const float* fc2_w  = (const float*)d_in[11];
  const float* fc2_b  = (const float*)d_in[12];
  const float* fc3_w  = (const float*)d_in[13];
  const float* fc3_b  = (const float*)d_in[14];

  // d_out regions double as live fp32 LSTM state buffers.
  float* out_y = (float*)d_out;               // [65536]
  float* h1    = out_y + NB * TT;             // [64][512]
  float* c1    = h1 + NB * H1;                // [64][512]
  float* h2    = c1 + NB * H1;                // [64][256]
  float* c2    = h2 + NB * H2W;               // [64][256]

  // Workspace bump allocator (256B aligned).
  char* ws = (char*)d_ws;
  size_t off = 0;
  auto alloc = [&](size_t bytes) -> char* {
    char* p = ws + off;
    off = (off + bytes + 255) & ~(size_t)255;
    return p;
  };
  _Float16* zero16  = (_Float16*)alloc((size_t)NB * H1 * 2);
  _Float16* x16     = (_Float16*)alloc((size_t)NB * TT * INW * 2);
  _Float16* wih1_16 = (_Float16*)alloc((size_t)4 * H1 * INW * 2);
  _Float16* whh1_16 = (_Float16*)alloc((size_t)4 * H1 * H1 * 2);
  _Float16* wih2_16 = (_Float16*)alloc((size_t)4 * H2W * H1 * 2);
  _Float16* whh2_16 = (_Float16*)alloc((size_t)4 * H2W * H2W * 2);
  _Float16* fc1w16  = (_Float16*)alloc((size_t)H4W * 512 * 2);
  _Float16* fc2w16  = (_Float16*)alloc((size_t)64 * H4W * 2);
  float*    bsum1   = (float*)alloc((size_t)4 * H1 * 4);
  float*    bsum2   = (float*)alloc((size_t)4 * H2W * 4);
  _Float16* hs1     = (_Float16*)alloc((size_t)NB * TT * H1 * 2);
  _Float16* hs2     = (_Float16*)alloc((size_t)NB * TT * H2W * 2);
  _Float16* ctx16   = (_Float16*)alloc((size_t)NB * TT * H2W * 2);
  _Float16* out1_16 = (_Float16*)alloc((size_t)NB * TT * H4W * 2);
  _Float16* out2_16 = (_Float16*)alloc((size_t)NB * TT * 64 * 2);

  auto cvt = [&](_Float16* dst, const float* src, int n) {
    k_f32_to_f16<<<(n + 255) / 256, 256, 0, stream>>>(dst, src, n);
  };

  // --- Phase 0: conversions / init -----------------------------------------
  cvt(x16, x, NB * TT * INW);
  cvt(wih1_16, W_ih1, 4 * H1 * INW);
  cvt(whh1_16, W_hh1, 4 * H1 * H1);
  cvt(wih2_16, W_ih2, 4 * H2W * H1);
  cvt(whh2_16, W_hh2, 4 * H2W * H2W);
  cvt(fc1w16, fc1_w, H4W * 512);
  k_pad_fc2<<<(64 * 128 + 255) / 256, 256, 0, stream>>>(fc2w16, fc2_w);
  k_vecadd<<<(4 * H1 + 255) / 256, 256, 0, stream>>>(bsum1, b_ih1, b_hh1, 4 * H1);
  k_vecadd<<<(4 * H2W + 255) / 256, 256, 0, stream>>>(bsum2, b_ih2, b_hh2, 4 * H2W);
  k_zero_f16<<<(NB * H1 + 255) / 256, 256, 0, stream>>>(zero16, NB * H1);
  k_zero_f32<<<(NB * H1 + 255) / 256, 256, 0, stream>>>(c1, NB * H1);
  k_zero_f32<<<(NB * H2W + 255) / 256, 256, 0, stream>>>(c2, NB * H2W);

  // --- Phase 1+2: two-layer LSTM, software-pipelined across launches --------
  // Launch k runs L1 step k and L2 step k-1 together (independent work).
  const int xP1 = TT * INW, hP1 = TT * H1, hP2 = TT * H2W;

  auto args1 = [&](int t) {
    LstmArgs a;
    a.xA = x16 + (size_t)t * INW;  a.xPitch = xP1;  a.KX = INW;
    a.hA = (t == 0) ? zero16 : (hs1 + (size_t)(t - 1) * H1);
    a.hPitch = (t == 0) ? H1 : hP1;  a.H = H1;
    a.Wx = wih1_16;  a.Wh = whh1_16;  a.bsum = bsum1;
    a.cbuf = c1;  a.hbuf32 = h1;
    a.hOut = hs1 + (size_t)t * H1;  a.hOutPitch = hP1;
    return a;
  };
  auto args2 = [&](int t) {
    LstmArgs a;
    a.xA = hs1 + (size_t)t * H1;   a.xPitch = hP1;  a.KX = H1;
    a.hA = (t == 0) ? zero16 : (hs2 + (size_t)(t - 1) * H2W);
    a.hPitch = (t == 0) ? H2W : hP2;  a.H = H2W;
    a.Wx = wih2_16;  a.Wh = whh2_16;  a.bsum = bsum2;
    a.cbuf = c2;  a.hbuf32 = h2;
    a.hOut = hs2 + (size_t)t * H2W;  a.hOutPitch = hP2;
    return a;
  };

  // L1 step 0 alone (16 blocks = 128 waves).
  {
    LstmArgs a0 = args1(0);
    k_lstm_pair<<<16, 256, 0, stream>>>(a0, 16, a0);
  }
  // Pipelined: L1 step t (16 blocks) + L2 step t-1 (8 blocks) in one launch.
  for (int t = 1; t < TT; ++t) {
    LstmArgs aL1 = args1(t), aL2 = args2(t - 1);
    k_lstm_pair<<<24, 256, 0, stream>>>(aL1, 16, aL2);
  }
  // L2 final step alone (8 blocks).
  {
    LstmArgs aF = args2(TT - 1);
    k_lstm_pair<<<8, 256, 0, stream>>>(aF, 8, aF);
  }

  // --- Phase 3: causal attention -------------------------------------------
  k_attn<<<NB * 8, 256, 0, stream>>>(hs2, ctx16);

  // --- Phase 4: FC head -----------------------------------------------------
  k_fc1<<<1024, 256, 0, stream>>>(ctx16, hs2, fc1w16, fc1_b, out1_16);
  k_fc2<<<512, 256, 0, stream>>>(out1_16, fc2w16, fc2_b, out2_16);
  k_fc3<<<(NB * TT + 255) / 256, 256, 0, stream>>>(out2_16, fc3_w, fc3_b, out_y);
}
